// KNN_31104153157973
// MI455X (gfx1250) — matrix operand order repaired and checked by reading, not compiled, not run on hardware
//
#include <hip/hip_runtime.h>
#include <hip/hip_bf16.h>

typedef __attribute__((ext_vector_type(16))) _Float16 v16h;
typedef __attribute__((ext_vector_type(8)))  _Float16 v8h;
typedef __attribute__((ext_vector_type(4)))  _Float16 v4h;
typedef __attribute__((ext_vector_type(8)))  float    v8f;
typedef __attribute__((ext_vector_type(4)))  float    v4f;

#define DDIM   512          // feature dim (fixed by reference)
#define NKC    (DDIM / 32)  // 16 k-chunks of 32 per WMMA
#define NSPLIT 125          // N-dimension splits (100000/125 = 800, multiple of 16)
#define KTOP   5

// ---------------------------------------------------------------------------
// lexicographic comparator matching jax.lax.top_k (value desc, index asc ties)
__device__ __forceinline__ bool better(float s, int i, float t, int j) {
  return (s > t) || (s == t && i < j);
}

// unrolled sorted-insert into a 5-deep register list (descending)
__device__ __forceinline__ void insert5(float s, int i,
    float& s0, int& i0, float& s1, int& i1, float& s2, int& i2,
    float& s3, int& i3, float& s4, int& i4) {
  if (!better(s, i, s4, i4)) return;
  if (better(s, i, s0, i0)) {
    s4=s3; i4=i3; s3=s2; i3=i2; s2=s1; i2=i1; s1=s0; i1=i0; s0=s; i0=i;
  } else if (better(s, i, s1, i1)) {
    s4=s3; i4=i3; s3=s2; i3=i2; s2=s1; i2=i1; s1=s; i1=i;
  } else if (better(s, i, s2, i2)) {
    s4=s3; i4=i3; s3=s2; i3=i2; s2=s; i2=i;
  } else if (better(s, i, s3, i3)) {
    s4=s3; i4=i3; s3=s; i3=i;
  } else {
    s4=s; i4=i;
  }
}

// ---------------------------------------------------------------------------
// One wave per row: f32 -> f16 conversion; optionally writes 0.5*||row||^2.
__global__ __launch_bounds__(256) void convert_rows(
    const float* __restrict__ src, _Float16* __restrict__ dst,
    float* __restrict__ halfsq, int nrows) {
  const int row  = (blockIdx.x * blockDim.x + threadIdx.x) >> 5;
  const int lane = threadIdx.x & 31;
  if (row >= nrows) return;
  const float* r = src + (size_t)row * DDIM;
  _Float16*    o = dst + (size_t)row * DDIM;
  float acc = 0.f;
#pragma unroll
  for (int j = 0; j < DDIM / 128; ++j) {       // 4 iters: (j*32+lane)*4
    const int c = (j * 32 + lane) * 4;
    v4f a = *(const v4f*)(r + c);
    acc += a[0]*a[0] + a[1]*a[1] + a[2]*a[2] + a[3]*a[3];
    v4h h; h[0] = (_Float16)a[0]; h[1] = (_Float16)a[1];
           h[2] = (_Float16)a[2]; h[3] = (_Float16)a[3];
    *(v4h*)(o + c) = h;
  }
  if (halfsq) {
#pragma unroll
    for (int off = 16; off > 0; off >>= 1) acc += __shfl_xor(acc, off, 32);
    if (lane == 0) halfsq[row] = 0.5f * acc;
  }
}

// ---------------------------------------------------------------------------
// Fused GEMM (f16 WMMA, f32 acc) + per-query top-5 over an N chunk.
// Wave w owns queries [blockIdx.x*128 + 16w, +16). All of K=512 for those
// queries is held in registers as 16 B-fragments (128 VGPRs).
// A fragment = 16 points x 32 k, loaded straight from row-major Xf16.
// C layout: lane L holds query (L&15), points (r + 8*(L>>4)) of the group.
__global__ __launch_bounds__(256) void knn_gemm_topk(
    const _Float16* __restrict__ Xh, const _Float16* __restrict__ Qh,
    const float* __restrict__ h2, float* __restrict__ candS,
    int* __restrict__ candI, int chunk, int N) {
  const int lane = threadIdx.x & 31;
  const int wave = threadIdx.x >> 5;
  const int col  = lane & 15;         // query slot / A-matrix row slot
  const int g    = lane >> 4;         // half-wave group
  const int q    = blockIdx.x * 128 + wave * 16 + col;

  // B fragments (queries): lane holds column N=col, k = kc*32 + 16g + [0,16)
  v16h qf[NKC];
  const _Float16* qrow = Qh + (size_t)q * DDIM;
#pragma unroll
  for (int kc = 0; kc < NKC; ++kc)
    qf[kc] = *(const v16h*)(qrow + kc * 32 + g * 16);

  const float NEG = -__builtin_inff();
  float s0 = NEG, s1 = NEG, s2 = NEG, s3 = NEG, s4 = NEG;
  int   i0 = 0x7fffffff, i1 = i0, i2 = i0, i3 = i0, i4 = i0;

  const int n0 = blockIdx.y * chunk;
  const int n1 = (n0 + chunk < N) ? (n0 + chunk) : N;   // multiples of 16

  for (int nb = n0; nb < n1; nb += 16) {
    const _Float16* xrow = Xh + (size_t)(nb + col) * DDIM;
    __builtin_prefetch(xrow + 16 * DDIM, 0, 1);   // next group's row -> global_prefetch
    v8f c = {0.f, 0.f, 0.f, 0.f, 0.f, 0.f, 0.f, 0.f};
#pragma unroll
    for (int kc = 0; kc < NKC; ++kc) {
      // A layout: lanes g=0: a[0..15]=k 0..15 ; g=1: k 16..31
      union { v16h v; v8h h[2]; } a;
      a.h[0] = *(const v8h*)(xrow + kc * 32 + g * 16);
      a.h[1] = *(const v8h*)(xrow + kc * 32 + g * 16 + 8);
      c = __builtin_amdgcn_wmma_f32_16x16x32_f16(
              false, a.v, false, qf[kc], (short)0, c, false, false);
    }
    // score = dot - 0.5*||x||^2 (monotone in neg_dist; q-norm is row-constant)
    v8f hh = *(const v8f*)(h2 + nb + 8 * g);
#pragma unroll
    for (int r = 0; r < 8; ++r) {
      const float s   = c[r] - hh[r];
      const int   idx = nb + 8 * g + r;
      insert5(s, idx, s0, i0, s1, i1, s2, i2, s3, i3, s4, i4);
    }
  }

  // lanes L and L^16 hold the same query over complementary points: merge.
  float t0 = __shfl_xor(s0, 16, 32), t1 = __shfl_xor(s1, 16, 32),
        t2 = __shfl_xor(s2, 16, 32), t3 = __shfl_xor(s3, 16, 32),
        t4 = __shfl_xor(s4, 16, 32);
  int   u0 = __shfl_xor(i0, 16, 32), u1 = __shfl_xor(i1, 16, 32),
        u2 = __shfl_xor(i2, 16, 32), u3 = __shfl_xor(i3, 16, 32),
        u4 = __shfl_xor(i4, 16, 32);
  insert5(t0, u0, s0, i0, s1, i1, s2, i2, s3, i3, s4, i4);
  insert5(t1, u1, s0, i0, s1, i1, s2, i2, s3, i3, s4, i4);
  insert5(t2, u2, s0, i0, s1, i1, s2, i2, s3, i3, s4, i4);
  insert5(t3, u3, s0, i0, s1, i1, s2, i2, s3, i3, s4, i4);
  insert5(t4, u4, s0, i0, s1, i1, s2, i2, s3, i3, s4, i4);

  if (g == 0) {
    const size_t base = ((size_t)q * NSPLIT + blockIdx.y) * KTOP;
    candS[base + 0] = s0; candI[base + 0] = i0;
    candS[base + 1] = s1; candI[base + 1] = i1;
    candS[base + 2] = s2; candI[base + 2] = i2;
    candS[base + 3] = s3; candI[base + 3] = i3;
    candS[base + 4] = s4; candI[base + 4] = i4;
  }
}

// ---------------------------------------------------------------------------
// Merge NSPLIT partial top-5 lists per query; vote = mean of 5 labels.
__global__ __launch_bounds__(256) void knn_finalize(
    const float* __restrict__ candS, const int* __restrict__ candI,
    const int* __restrict__ Y, float* __restrict__ out, int Bq) {
  const int qi = blockIdx.x * blockDim.x + threadIdx.x;
  if (qi >= Bq) return;
  const float NEG = -__builtin_inff();
  float s0 = NEG, s1 = NEG, s2 = NEG, s3 = NEG, s4 = NEG;
  int   i0 = 0x7fffffff, i1 = i0, i2 = i0, i3 = i0, i4 = i0;
  const float* cs = candS + (size_t)qi * NSPLIT * KTOP;
  const int*   ci = candI + (size_t)qi * NSPLIT * KTOP;
  for (int t = 0; t < NSPLIT * KTOP; ++t)
    insert5(cs[t], ci[t], s0, i0, s1, i1, s2, i2, s3, i3, s4, i4);
  const float votes =
      (float)(Y[i0] + Y[i1] + Y[i2] + Y[i3] + Y[i4]) * 0.2f;
  out[2 * qi + 0] = votes;
  out[2 * qi + 1] = 0.f;     // pad column
}

// ---------------------------------------------------------------------------
extern "C" void kernel_launch(void* const* d_in, const int* in_sizes, int n_in,
                              void* d_out, int out_size, void* d_ws, size_t ws_size,
                              hipStream_t stream) {
  const float* Q = (const float*)d_in[0];   // [B, 512]
  const float* X = (const float*)d_in[1];   // [N, 512]
  const int*   Y = (const int*)d_in[2];     // [N] labels in {0,1}
  // d_in[3] = K (fixed 5 by setup)

  const int Bq = in_sizes[0] / DDIM;        // 2048
  const int N  = in_sizes[2];               // 100000

  // workspace carve (256B aligned): Xf16 102.4MB | Qf16 2MB | 0.5*x2 | cands
  size_t off = 0;
  auto carve = [&](size_t bytes) -> void* {
    void* p = (char*)d_ws + off;
    off += (bytes + 255) & ~(size_t)255;
    return p;
  };
  _Float16* Xh    = (_Float16*)carve((size_t)N  * DDIM * sizeof(_Float16));
  _Float16* Qh    = (_Float16*)carve((size_t)Bq * DDIM * sizeof(_Float16));
  float*    h2    = (float*)   carve((size_t)N * sizeof(float));
  float*    candS = (float*)   carve((size_t)Bq * NSPLIT * KTOP * sizeof(float));
  int*      candI = (int*)     carve((size_t)Bq * NSPLIT * KTOP * sizeof(int));
  (void)ws_size;

  // chunk per N-split, rounded up to a 16-point group (800 for N=100000)
  int chunk = (N + NSPLIT - 1) / NSPLIT;
  chunk = (chunk + 15) & ~15;

  convert_rows<<<dim3((N  + 7) / 8), 256, 0, stream>>>(X, Xh, h2, N);
  convert_rows<<<dim3((Bq + 7) / 8), 256, 0, stream>>>(Q, Qh, nullptr, Bq);

  knn_gemm_topk<<<dim3(Bq / 128, NSPLIT), 256, 0, stream>>>(
      Xh, Qh, h2, candS, candI, chunk, N);

  knn_finalize<<<dim3((Bq + 255) / 256), 256, 0, stream>>>(
      candS, candI, Y, (float*)d_out, Bq);
}
